// SteerableFirstLayer_30459908063681
// MI455X (gfx1250) — matrix operand order repaired and checked by reading, not compiled, not run on hardware
//
#include <hip/hip_runtime.h>
#include <hip/hip_bf16.h>

typedef __attribute__((ext_vector_type(16))) _Float16 v16h;
typedef __attribute__((ext_vector_type(8)))  _Float16 v8h;
typedef __attribute__((ext_vector_type(8)))  float    v8f;

#define B_   32
#define OC_  64
#define IC_  2
#define KW_  65
#define L_   65536
#define KPC  80            // per-channel padded K (65 taps + 15 zero-weight taps)
#define KPAD (IC_ * KPC)   // 160 = 5 chunks of 32
#define NPAD (2 * KW_)     // 130
#define LTILE 256
#define SLOT 352           // per-channel x slice length (>= 350), mult of 8
#define ROWL 736           // replicated pool row length (2*SLOT + shift + tail), mult of 8
#define FS_  50000000.0f
#define PI_F 3.14159265358979f

union V16U { v16h v; v8h h[2]; };

__device__ __forceinline__ float nan_to_num(float v) {
    if (isnan(v)) return 0.0f;
    if (isinf(v)) return v > 0.0f ? 3.4028234663852886e38f : -3.4028234663852886e38f;
    return v;
}

// ---------------------------------------------------------------------------
// Kernel 1: steered-weight preparation (f32 math, f16 output into d_ws).
// M = U @ D has <=4 non-zeros per row (two chained linear interps), so each
// steered tap is a 4-tap weighted read of base_w. Shift/phase/gain resolved
// at the source tap p = (i - sh) mod 65. Padded taps 65..79 are zeroed so the
// GEMM kernel needs no masking.
// ---------------------------------------------------------------------------
__global__ __launch_bounds__(256)
void steer_weights_kernel(const int* __restrict__ z, const float* __restrict__ s,
                          const float* __restrict__ base_w, _Float16* __restrict__ Wh) {
    const int b = blockIdx.x;

    float sf[5];
    int   zb[5];
#pragma unroll
    for (int i = 0; i < 5; ++i) { sf[i] = nan_to_num(s[b * 5 + i]); zb[i] = z[b * 5 + i]; }
    const float f0    = (zb[0] > 0) ? sf[0] : 0.0f;
    const float alpha = (zb[1] > 0) ? sf[1] : 1.0f;
    const float gain  = (zb[2] > 0) ? sf[2] : 1.0f;
    const float shf   = (zb[3] > 0) ? sf[3] : 0.0f;
    const float chirp = (zb[4] > 0) ? sf[4] : 0.0f;

    const float a  = fmaxf(alpha, 0.001f);
    float Nr = rintf((float)KW_ / a);
    Nr = fminf(fmaxf(Nr, 1.0f), (float)NPAD);
    const int   N  = (int)Nr;
    const float Nf = (float)N;
    const int   sh = (int)rintf(shf);

    _Float16* whb = Wh + (size_t)b * OC_ * KPAD;

    for (int idx = threadIdx.x; idx < OC_ * KW_; idx += blockDim.x) {
        const int oc = idx / KW_;
        const int i  = idx % KW_;                       // output tap after circular shift
        const int p  = ((i - sh) % KW_ + KW_) % KW_;    // source tap

        const float c2 = fmaxf(((float)p + 0.5f) * (Nf / (float)KW_) - 0.5f, 0.0f);
        const int   j0 = (int)floorf(c2);
        const float w2 = c2 - (float)j0;
        const int   j1 = min(j0 + 1, N - 1);

        const float* bI = base_w + (size_t)oc * IC_ * KW_;   // ic = 0
        const float* bQ = bI + KW_;                          // ic = 1

        float accI = 0.0f, accQ = 0.0f;
#pragma unroll
        for (int t = 0; t < 2; ++t) {
            const int   j  = (t == 0) ? j0 : j1;
            const float uw = (t == 0) ? (1.0f - w2) : w2;
            if (j < N) {
                const float c   = fmaxf(((float)j + 0.5f) * ((float)KW_ / Nf) - 0.5f, 0.0f);
                const float i0f = floorf(c);
                const float w1  = c - i0f;
                int i0 = (int)i0f; i0 = min(max(i0, 0), KW_ - 1);
                const int i1 = min(i0 + 1, KW_ - 1);
                accI += uw * ((1.0f - w1) * bI[i0] + w1 * bI[i1]);
                accQ += uw * ((1.0f - w1) * bQ[i0] + w1 * bQ[i1]);
            }
        }

        const float tt  = (float)p / FS_;
        const float phi = 2.0f * PI_F * f0 * tt + PI_F * chirp * tt * tt;
        const float cph = __cosf(phi), sph = __sinf(phi);
        const float Ip = accI * cph - accQ * sph;
        const float Qp = accI * sph + accQ * cph;

        whb[oc * KPAD + 0 * KPC + i] = (_Float16)(gain * Ip);
        whb[oc * KPAD + 1 * KPC + i] = (_Float16)(gain * Qp);
    }

    // Zero the padded taps (65..79, both channels) -> GEMM needs no masks.
    for (int t = threadIdx.x; t < OC_ * 2 * (KPC - KW_); t += blockDim.x) {
        const int oc = t / (2 * (KPC - KW_));
        const int r  = t % (2 * (KPC - KW_));
        const int ic = r / (KPC - KW_);
        const int kk = KW_ + (r % (KPC - KW_));
        whb[oc * KPAD + ic * KPC + kk] = (_Float16)0;
    }
}

// ---------------------------------------------------------------------------
// Kernel 2: per-batch GEMM (M=64, K=160, N=65536) via v_wmma_f32_16x16x32_f16.
// Block = 8 waves per (batch, 256-column) tile. Each wave owns 32 distinct
// columns (2 WMMA col-subtiles) x all 4 OC tiles -> zero redundant B builds.
// B operands: a lane's 16 values per K-chunk are 16 CONSECUTIVE LDS halfs
// (zero-weight K padding removes masking), fetched as two aligned
// ds_load_b128 from an 8-way shift-replicated x pool.
// ---------------------------------------------------------------------------
__global__ __launch_bounds__(256)
void conv_wmma_kernel(const float* __restrict__ x, const _Float16* __restrict__ Wh,
                      float* __restrict__ y) {
    const int b  = blockIdx.y;
    const int l0 = blockIdx.x * LTILE;

    __shared__ __align__(16) _Float16 wsl[OC_ * KPAD];   // 20 KB steered weights
    __shared__ __align__(16) _Float16 xpool[8 * ROWL];   // 11.5 KB shifted x copies

    // Stage steered weights for this batch (contiguous f16 -> wide loads).
    for (int t = threadIdx.x; t < OC_ * KPAD; t += blockDim.x)
        wsl[t] = Wh[(size_t)b * OC_ * KPAD + t];

    // Stage 8 shifted copies of the padded x slice.
    // Logical q in [0,704): ic = q/SLOT, tt = q%SLOT; tt<320 maps to
    // x[b, ic, l0-32+tt]; everything else (halo, slot tail) is zero.
    // Copy r stores data[q] at physical r*ROWL + q + r.
#pragma unroll
    for (int r = 0; r < 8; ++r) {
        for (int p = threadIdx.x; p < ROWL; p += blockDim.x) {
            const int q = p - r;
            float v = 0.0f;
            if (q >= 0 && q < 2 * SLOT) {
                const int ic = (q >= SLOT) ? 1 : 0;
                const int tt = q - ic * SLOT;
                if (tt < LTILE + 64) {
                    const long gl = (long)l0 - 32 + tt;
                    if (gl >= 0 && gl < (long)L_)
                        v = x[((size_t)b * IC_ + ic) * L_ + gl];
                }
            }
            xpool[r * ROWL + p] = (_Float16)v;
        }
    }
    __syncthreads();

    const int lane = threadIdx.x & 31;
    const int wave = threadIdx.x >> 5;
    const int cw   = wave * 32;            // this wave's 32 distinct columns
    const int n    = lane & 15;            // B/N-lane: column within subtile
    const int mlo  = lane & 15;            // A/M-lane: row within OC tile
    const int akb  = (lane < 16) ? 0 : 16; // A K-offset group (contiguous, matches B)
    const int bko  = (lane < 16) ? 0 : 16; // B K-offset group (contiguous, matches A)

#pragma unroll
    for (int ct = 0; ct < 2; ++ct) {
        const int c0 = cw + ct * 16;

        // Build B[5 chunks] once; reused by all 4 OC tiles.
        v16h Bv[5];
#pragma unroll
        for (int kc = 0; kc < 5; ++kc) {
            const int kbase = kc * 32 + bko;               // segment start in K
            const int ic    = (kbase >= KPC) ? 1 : 0;
            const int kk0   = kbase - ic * KPC;
            const int q     = ic * SLOT + kk0 + c0 + n;    // 16 consecutive halfs
            const int r     = (-q) & 7;                    // shifted copy -> 16B aligned
            const v8h* pa   = (const v8h*)&xpool[r * ROWL + q + r];
            V16U t; t.h[0] = pa[0]; t.h[1] = pa[1];
            Bv[kc] = t.v;
        }

#pragma unroll
        for (int oct = 0; oct < 4; ++oct) {
            const int m = oct * 16 + mlo;
            v8f acc = {};
#pragma unroll
            for (int kc = 0; kc < 5; ++kc) {
                // A 16-bit 16x32 layout: halfs {kb..kb+15} (contiguous, same as B),
                // two 16B-aligned runs in the weight row.
                const v8h* wa = (const v8h*)&wsl[m * KPAD + kc * 32 + akb];
                V16U t; t.h[0] = wa[0]; t.h[1] = wa[1];
                acc = __builtin_amdgcn_wmma_f32_16x16x32_f16(
                    false, t.v, false, Bv[kc], (short)0, acc, false, false);
            }
            // C/D layout: VGPR rr -> (M = rr + (lane>=16 ? 8:0), N = lane&15).
            const int    ocb     = oct * 16 + ((lane >= 16) ? 8 : 0);
            const size_t colBase = (size_t)l0 + c0 + n;
#pragma unroll
            for (int rr = 0; rr < 8; ++rr)
                y[((size_t)b * OC_ + ocb + rr) * L_ + colBase] = acc[rr];
        }
    }
}

extern "C" void kernel_launch(void* const* d_in, const int* in_sizes, int n_in,
                              void* d_out, int out_size, void* d_ws, size_t ws_size,
                              hipStream_t stream) {
    const float* x      = (const float*)d_in[0];   // (32, 2, 1, 65536) f32
    const int*   z      = (const int*)  d_in[1];   // (32, 5) i32
    const float* s      = (const float*)d_in[2];   // (32, 5) f32
    const float* base_w = (const float*)d_in[3];   // (64, 2, 1, 65) f32
    float*       y      = (float*)d_out;           // (32, 64, 1, 65536) f32
    _Float16*    Wh     = (_Float16*)d_ws;         // 32*64*160 f16 = 640 KB scratch

    steer_weights_kernel<<<B_, 256, 0, stream>>>(z, s, base_w, Wh);

    dim3 grid(L_ / LTILE, B_);
    conv_wmma_kernel<<<grid, 256, 0, stream>>>(x, Wh, y);
}
